// GeoAttention_28278064676911
// MI455X (gfx1250) — matrix operand order repaired and checked by reading, not compiled, not run on hardware
//
#include <hip/hip_runtime.h>

typedef __attribute__((ext_vector_type(16))) _Float16 v16h;
typedef __attribute__((ext_vector_type(8)))  float    v8f;
typedef __attribute__((ext_vector_type(4)))  unsigned int u32x4;
typedef __attribute__((ext_vector_type(8)))  unsigned int u32x8;

#define DEV __device__ __forceinline__

union HF16 { v16h v; uint4 u[2]; };   // one WMMA 16-bit fragment (32B)
union HU4  { uint4 u; _Float16 h[8]; };

DEV v8f vzero8() { v8f z = {0.f,0.f,0.f,0.f,0.f,0.f,0.f,0.f}; return z; }

DEV v8f wmma16(v16h a, v16h b, v8f c) {
  // D = A(16x32 f16) * B(32x16 f16) + C(16x16 f32)
  return __builtin_amdgcn_wmma_f32_16x16x32_f16(false, a, false, b, (short)0, c, false, false);
}

// reductions across the 16 lanes that share one C-tile row group
DEV float wmax16(float v) {
  v = fmaxf(v, __shfl_xor(v, 1, 32));
  v = fmaxf(v, __shfl_xor(v, 2, 32));
  v = fmaxf(v, __shfl_xor(v, 4, 32));
  v = fmaxf(v, __shfl_xor(v, 8, 32));
  return v;
}
DEV float wsum16(float v) {
  v += __shfl_xor(v, 1, 32);
  v += __shfl_xor(v, 2, 32);
  v += __shfl_xor(v, 4, 32);
  v += __shfl_xor(v, 8, 32);
  return v;
}

// Issue a 1-D Tensor Data Mover copy: nelem f16 elements, global -> LDS.
// D# per CDNA5 ISA 8.3/8.4: group0 = {count, lds_addr, global_addr, type=2},
// group1 = {data_size=2B, tensor_dim0=tile_dim0=nelem, stride=nelem}.
DEV void tdm_load_1d(unsigned lds_off, const void* gaddr, unsigned nelem) {
  unsigned long long ga = (unsigned long long)gaddr;
  u32x4 g0;
  g0[0] = 1u;                                   // count=1, user mode
  g0[1] = lds_off;                              // lds_addr (bytes)
  g0[2] = (unsigned)ga;                         // global_addr[31:0]
  g0[3] = (unsigned)(ga >> 32) | (2u << 30);    // global_addr[56:32] | type=2
  u32x8 g1;
  g1[0] = (1u << 16);                           // data_size=1 (2 bytes/elem)
  g1[1] = (nelem & 0xFFFFu) << 16;              // tensor_dim0[15:0]
  g1[2] = (nelem >> 16) | (1u << 16);           // tensor_dim0[31:16] | tensor_dim1=1
  g1[3] = (nelem & 0xFFFFu) << 16;              // tile_dim0
  g1[4] = 0u;                                   // tile_dim1=0, tile_dim2=0 (1-D)
  g1[5] = nelem;                                // tensor_dim0_stride[31:0]
  g1[6] = 0u;
  g1[7] = 0u;
  u32x4 g2 = {0u, 0u, 0u, 0u};                  // dims 3/4 unused (tile_dim3=0)
  u32x4 g3 = {0u, 0u, 0u, 0u};
  asm volatile("tensor_load_to_lds %0, %1, %2, %3"
               :: "s"(g0), "s"(g1), "s"(g2), "s"(g3)
               : "memory");
}

// ---- weight conversion: q/k/v_proj [8,512,64] fp32 -> f16 transposed [h][k][d]
__global__ __launch_bounds__(256) void geo_cvt_projw(
    const float* __restrict__ qp, const float* __restrict__ kp, const float* __restrict__ vp,
    _Float16* __restrict__ wq, _Float16* __restrict__ wk, _Float16* __restrict__ wv) {
  int idx = blockIdx.x * 256 + threadIdx.x;      // 0..786431
  int w   = idx >> 18;                           // 262144 elems per weight
  int rem = idx & 262143;
  int h   = rem >> 15;                           // 32768 per head
  int r2  = rem & 32767;
  int k   = r2 >> 9;
  int d   = r2 & 511;
  const float* src = (w == 0) ? qp : (w == 1) ? kp : vp;
  _Float16*    dst = (w == 0) ? wq : (w == 1) ? wk : wv;
  dst[rem] = (_Float16)src[(h * 512 + d) * 64 + k];   // [h][k][d]
}

// ---- out_w [512,512] fp32 -> f16 (kept [o][f]: already B-fragment friendly)
__global__ __launch_bounds__(256) void geo_cvt_outw(const float* __restrict__ ow,
                                                    _Float16* __restrict__ wo) {
  int idx = blockIdx.x * 256 + threadIdx.x;      // 0..262143
  wo[idx] = (_Float16)ow[idx];
}

// ---- fused per-head Q/K/V projection GEMM: X[4096,512] x W[512,64] (+bias, Q*=0.125)
__global__ __launch_bounds__(256) void geo_proj(
    const float* __restrict__ q, const float* __restrict__ k, const float* __restrict__ v,
    const float* __restrict__ qb, const float* __restrict__ kb, const float* __restrict__ vb,
    const _Float16* __restrict__ wq, const _Float16* __restrict__ wk, const _Float16* __restrict__ wv,
    _Float16* __restrict__ Qp, _Float16* __restrict__ Kp, _Float16* __restrict__ Vp) {
  __shared__ __align__(16) _Float16 Al[128 * 32];
  __shared__ __align__(16) _Float16 Bl[64 * 32];
  const int z = blockIdx.z;
  const float*    X    = (z == 0) ? q  : (z == 1) ? k  : v;
  const float*    bias = (z == 0) ? qb : (z == 1) ? kb : vb;
  const _Float16* Wt   = (z == 0) ? wq : (z == 1) ? wk : wv;
  _Float16*       Out  = (z == 0) ? Qp : (z == 1) ? Kp : Vp;
  const float oscale = (z == 0) ? 0.125f : 1.0f;   // fold 1/sqrt(d_k) into Q
  const int h = blockIdx.y;
  const int Rbase = blockIdx.x * 128;
  const int t = threadIdx.x;
  const int wv32 = t >> 5, L = t & 31, l16 = L & 15, hi = L >> 4;

  v8f acc[4];
  #pragma unroll
  for (int c = 0; c < 4; ++c) acc[c] = vzero8();

  for (int kk = 0; kk < 16; ++kk) {
    const int d0 = kk * 32;
    { // stage A tile 128x32 (fp32 -> f16)
      int row = t >> 1, half = t & 1;
      const float* srow = X + (size_t)(Rbase + row) * 512 + d0 + half * 16;
      HU4 pk0, pk1;
      #pragma unroll
      for (int i = 0; i < 4; ++i) {
        float4 f = ((const float4*)srow)[i];
        HU4& dst = (i < 2) ? pk0 : pk1;
        int o = (i & 1) * 4;
        dst.h[o+0] = (_Float16)f.x; dst.h[o+1] = (_Float16)f.y;
        dst.h[o+2] = (_Float16)f.z; dst.h[o+3] = (_Float16)f.w;
      }
      *(uint4*)&Al[row * 32 + half * 16]     = pk0.u;
      *(uint4*)&Al[row * 32 + half * 16 + 8] = pk1.u;
    }
    { // stage B tile: Bl[kcol][dd] = Wt[h][kcol][d0+dd]
      int kcol = t >> 2, part = t & 3;
      *(uint4*)&Bl[kcol * 32 + part * 8] =
          *(const uint4*)(Wt + (size_t)h * 32768 + (size_t)kcol * 512 + d0 + part * 8);
    }
    __syncthreads();
    HF16 a;
    const _Float16* ar = &Al[(wv32 * 16 + l16) * 32];
    a.u[0] = *(const uint4*)(ar + hi * 8);
    a.u[1] = *(const uint4*)(ar + hi * 8 + 16);
    #pragma unroll
    for (int c = 0; c < 4; ++c) {
      HF16 bf;
      const _Float16* br = &Bl[(c * 16 + l16) * 32 + hi * 8];
      bf.u[0] = *(const uint4*)(br);
      bf.u[1] = *(const uint4*)(br + 16);
      acc[c] = wmma16(a.v, bf.v, acc[c]);
    }
    __syncthreads();
  }

  #pragma unroll
  for (int c = 0; c < 4; ++c) {
    float bv = bias[h * 64 + c * 16 + l16];
    #pragma unroll
    for (int r = 0; r < 8; ++r) {
      int R = Rbase + wv32 * 16 + r + hi * 8;
      int b_ = R >> 11, n = R & 2047;
      float val = (acc[c][r] + bv) * oscale;
      Out[((size_t)(b_ * 8 + h) * 2048 + n) * 64 + c * 16 + l16] = (_Float16)val;
    }
  }
}

// dynamic-LDS layout for geo_attn (dynamic segment starts at LDS offset 0)
#define OFF_K   0u        // K tile  [32 key][64 dk] f16    (4 KB)
#define OFF_VR  4096u     // V tile  [32 key][64 dk] f16    (4 KB, row major)
#define OFF_VT  8192u     // V tile  [64 dk][32 key] f16    (4 KB, transposed)
#define OFF_CK  12288u    // key coords + mask bias float4[32] (512 B)
#define OFF_P   12800u    // per-wave P relayout scratch 8*16*32 f16 (8 KB)
#define SMEM_BYTES 20992u

// ---- flash attention with geometric RBF bias + key padding mask
__global__ __launch_bounds__(256) void geo_attn(
    const _Float16* __restrict__ Qp, const _Float16* __restrict__ Kp,
    const _Float16* __restrict__ Vp, const float* __restrict__ coords,
    const unsigned char* __restrict__ mask, _Float16* __restrict__ Ob) {
  extern __shared__ __align__(16) char smem[];
  _Float16* Kl = (_Float16*)(smem + OFF_K);
  _Float16* Vr = (_Float16*)(smem + OFF_VR);
  _Float16* Vt = (_Float16*)(smem + OFF_VT);
  float4*   Ck = (float4*)(smem + OFF_CK);
  _Float16* Pl = (_Float16*)(smem + OFF_P);

  const int h = blockIdx.y, b = blockIdx.z;
  const int qbase = blockIdx.x * 128;
  const int t = threadIdx.x;
  const int w = t >> 5, L = t & 31, l16 = L & 15, hi = L >> 4;
  const size_t bh = (size_t)(b * 8 + h) * 2048;

  float ls = __powf(20.0f, (float)h * (1.0f / 7.0f));
  float spread = 1.0f + 5.0f * (ls - 1.0f) / 19.0f;
  const float inv2s2 = 1.0f / (2.0f * spread * spread);

  // per-wave Q fragments (Q already scaled by 1/sqrt(d_k))
  const _Float16* qrow = Qp + (bh + qbase + w * 16 + l16) * 64;
  HF16 q0, q1;
  q0.u[0] = *(const uint4*)(qrow + hi * 8);
  q0.u[1] = *(const uint4*)(qrow + hi * 8 + 16);
  q1.u[0] = *(const uint4*)(qrow + 32 + hi * 8);
  q1.u[1] = *(const uint4*)(qrow + 32 + hi * 8 + 16);

  float cqx[8], cqy[8], cqz[8];
  #pragma unroll
  for (int r = 0; r < 8; ++r) {
    int qg = qbase + w * 16 + r + hi * 8;
    const float* cp = coords + ((size_t)b * 2048 + qg) * 3;
    cqx[r] = cp[0]; cqy[r] = cp[1]; cqz[r] = cp[2];
  }

  v8f accO[4];
  #pragma unroll
  for (int c = 0; c < 4; ++c) accO[c] = vzero8();
  float rmax[8], rsum[8];
  #pragma unroll
  for (int r = 0; r < 8; ++r) { rmax[r] = -3.0e38f; rsum[r] = 0.0f; }

  for (int kt = 0; kt < 64; ++kt) {
    const int kb = kt * 32;
    // TDM: DMA the K and V tiles (each 2048 f16, fully contiguous) into LDS.
    // Wave 0 issues both descriptors and waits TENSORcnt; barrier publishes.
    if (w == 0) {
      tdm_load_1d(OFF_K,  Kp + (bh + kb) * 64, 2048u);
      tdm_load_1d(OFF_VR, Vp + (bh + kb) * 64, 2048u);
    }
    if (t < 32) { // key coords + mask bias (wave 0, overlaps with TDM)
      int j = kb + t;
      const float* cp = coords + ((size_t)b * 2048 + j) * 3;
      float mb = mask[(size_t)b * 2048 + j] ? 0.0f : -1.0e9f;
      Ck[t] = make_float4(cp[0], cp[1], cp[2], mb);
    }
    if (kt + 1 < 64) {  // pull next K/V tile toward L2 (global_prefetch_b8)
      __builtin_prefetch(Kp + (bh + kb + 32 + (t >> 3)) * 64, 0, 1);
      __builtin_prefetch(Vp + (bh + kb + 32 + (t >> 3)) * 64, 0, 1);
    }
    if (w == 0) __builtin_amdgcn_s_wait_tensorcnt(0);
    __syncthreads();

    { // LDS->LDS transpose of V for the P*V B-fragment (key-major)
      int key = t >> 3, dkb = (t & 7) * 8;
      HU4 tv; tv.u = *(const uint4*)(Vr + key * 64 + dkb);
      #pragma unroll
      for (int j = 0; j < 8; ++j) Vt[(dkb + j) * 32 + key] = tv.h[j];
    }
    __syncthreads();

    // S = Q K^T + bias + mask  (two 16-key subtiles)
    v8f s[2];
    #pragma unroll
    for (int st = 0; st < 2; ++st) {
      HF16 b0, b1;
      const _Float16* kr = &Kl[(st * 16 + l16) * 64];
      b0.u[0] = *(const uint4*)(kr + hi * 8);
      b0.u[1] = *(const uint4*)(kr + hi * 8 + 16);
      b1.u[0] = *(const uint4*)(kr + 32 + hi * 8);
      b1.u[1] = *(const uint4*)(kr + 32 + hi * 8 + 16);
      v8f zacc = vzero8();
      zacc = wmma16(q0.v, b0.v, zacc);
      zacc = wmma16(q1.v, b1.v, zacc);
      float4 kc = Ck[st * 16 + l16];
      #pragma unroll
      for (int r = 0; r < 8; ++r) {
        float dx = cqx[r] - kc.x, dy = cqy[r] - kc.y, dz = cqz[r] - kc.z;
        zacc[r] += kc.w - (dx * dx + dy * dy + dz * dz) * inv2s2;
      }
      s[st] = zacc;
    }

    // online softmax + write P (C-layout -> A-layout via per-wave LDS)
    _Float16* Pw = &Pl[w * 16 * 32];
    #pragma unroll
    for (int r = 0; r < 8; ++r) {
      float tm = wmax16(fmaxf(s[0][r], s[1][r]));
      float nm = fmaxf(rmax[r], tm);
      float alpha = __expf(rmax[r] - nm);
      rmax[r] = nm;
      float e0 = __expf(s[0][r] - nm);
      float e1 = __expf(s[1][r] - nm);
      rsum[r] = rsum[r] * alpha + wsum16(e0 + e1);
      #pragma unroll
      for (int c = 0; c < 4; ++c) accO[c][r] *= alpha;
      Pw[(r + hi * 8) * 32 + l16]      = (_Float16)e0;
      Pw[(r + hi * 8) * 32 + 16 + l16] = (_Float16)e1;
    }
    asm volatile("s_wait_dscnt 0" ::: "memory");  // wave-private LDS RAW

    // O += P * V
    HF16 pa;
    const _Float16* pr = &Pw[l16 * 32];
    pa.u[0] = *(const uint4*)(pr + hi * 8);
    pa.u[1] = *(const uint4*)(pr + hi * 8 + 16);
    #pragma unroll
    for (int c = 0; c < 4; ++c) {
      HF16 vb;
      const _Float16* vr = &Vt[(c * 16 + l16) * 32 + hi * 8];
      vb.u[0] = *(const uint4*)(vr);
      vb.u[1] = *(const uint4*)(vr + 16);
      accO[c] = wmma16(pa.v, vb.v, accO[c]);
    }
    __syncthreads();
  }

  // normalize and write in torch permute(0,2,3,1) order: feature = k*8 + h
  #pragma unroll
  for (int r = 0; r < 8; ++r) {
    float inv = 1.0f / rsum[r];
    int qg = qbase + w * 16 + r + hi * 8;
    _Float16* orow = Ob + ((size_t)b * 2048 + qg) * 512 + h;
    #pragma unroll
    for (int c = 0; c < 4; ++c)
      orow[(size_t)(c * 16 + l16) * 8] = (_Float16)(accO[c][r] * inv);
  }
}

// ---- output projection: out = Ob[4096,512] @ out_w^T + out_b (fp32 out)
__global__ __launch_bounds__(256) void geo_out(
    const _Float16* __restrict__ Ob, const _Float16* __restrict__ Wo,
    const float* __restrict__ ob, float* __restrict__ out) {
  __shared__ __align__(16) _Float16 Al[128 * 32];
  __shared__ __align__(16) _Float16 Bl[64 * 32];
  const int Rbase = blockIdx.x * 128;
  const int obase = blockIdx.y * 64;
  const int t = threadIdx.x;
  const int w = t >> 5, L = t & 31, l16 = L & 15, hi = L >> 4;

  v8f acc[4];
  #pragma unroll
  for (int c = 0; c < 4; ++c) acc[c] = vzero8();

  for (int kk = 0; kk < 16; ++kk) {
    const int f0 = kk * 32;
    {
      int row = t >> 1, half = t & 1;
      const uint4* s = (const uint4*)(Ob + (size_t)(Rbase + row) * 512 + f0 + half * 16);
      *(uint4*)&Al[row * 32 + half * 16]     = s[0];
      *(uint4*)&Al[row * 32 + half * 16 + 8] = s[1];
    }
    {
      int oc = t >> 2, part = t & 3;
      *(uint4*)&Bl[oc * 32 + part * 8] =
          *(const uint4*)(Wo + (size_t)(obase + oc) * 512 + f0 + part * 8);
    }
    __syncthreads();
    HF16 a;
    const _Float16* ar = &Al[(w * 16 + l16) * 32];
    a.u[0] = *(const uint4*)(ar + hi * 8);
    a.u[1] = *(const uint4*)(ar + hi * 8 + 16);
    #pragma unroll
    for (int c = 0; c < 4; ++c) {
      HF16 bf;
      const _Float16* br = &Bl[(c * 16 + l16) * 32 + hi * 8];
      bf.u[0] = *(const uint4*)(br);
      bf.u[1] = *(const uint4*)(br + 16);
      acc[c] = wmma16(a.v, bf.v, acc[c]);
    }
    __syncthreads();
  }

  #pragma unroll
  for (int c = 0; c < 4; ++c) {
    int o = obase + c * 16 + l16;
    float bv = ob[o];
    #pragma unroll
    for (int r = 0; r < 8; ++r) {
      int R = Rbase + w * 16 + r + hi * 8;
      out[(size_t)R * 512 + o] = acc[c][r] + bv;
    }
  }
}

extern "C" void kernel_launch(void* const* d_in, const int* in_sizes, int n_in,
                              void* d_out, int out_size, void* d_ws, size_t ws_size,
                              hipStream_t stream) {
  (void)in_sizes; (void)n_in; (void)out_size; (void)ws_size;
  const float* in_q      = (const float*)d_in[0];
  const float* in_k      = (const float*)d_in[1];
  const float* in_v      = (const float*)d_in[2];
  const float* in_coords = (const float*)d_in[3];
  const unsigned char* in_mask = (const unsigned char*)d_in[4];
  const float* in_qproj  = (const float*)d_in[5];
  const float* in_kproj  = (const float*)d_in[6];
  const float* in_vproj  = (const float*)d_in[7];
  const float* in_qbias  = (const float*)d_in[8];
  const float* in_kbias  = (const float*)d_in[9];
  const float* in_vbias  = (const float*)d_in[10];
  const float* in_outw   = (const float*)d_in[11];
  const float* in_outb   = (const float*)d_in[12];
  float* out = (float*)d_out;

  // workspace layout (f16): 4x 262144 weights + 4x 2097152 activations = 18 MB
  _Float16* WqT = (_Float16*)d_ws;
  _Float16* WkT = WqT + 262144;
  _Float16* WvT = WkT + 262144;
  _Float16* Wo  = WvT + 262144;
  _Float16* Qp  = Wo  + 262144;
  _Float16* Kp  = Qp  + 2097152;
  _Float16* Vp  = Kp  + 2097152;
  _Float16* Ob  = Vp  + 2097152;

  geo_cvt_projw<<<3072, 256, 0, stream>>>(in_qproj, in_kproj, in_vproj, WqT, WkT, WvT);
  geo_cvt_outw<<<1024, 256, 0, stream>>>(in_outw, Wo);
  geo_proj<<<dim3(32, 8, 3), 256, 0, stream>>>(in_q, in_k, in_v,
                                               in_qbias, in_kbias, in_vbias,
                                               WqT, WkT, WvT, Qp, Kp, Vp);
  geo_attn<<<dim3(16, 8, 2), 256, SMEM_BYTES, stream>>>(Qp, Kp, Vp, in_coords, in_mask, Ob);
  geo_out<<<dim3(32, 8), 256, 0, stream>>>(Ob, Wo, in_outb, out);
}
